// PhysicsAttention_43860206027247
// MI455X (gfx1250) — compile-verified, matrix operand order checked by reading, not run on hardware
//
#include <hip/hip_runtime.h>

#define HH 8
#define DD 64
#define GG 64
#define HID2 512
#define S1 64          // row splits for pass 1
#define R3 64          // row splits for pass 3
#define EPSF 1e-5f
#define LSTR 520       // f16 LDS operand row stride (512 + 8 pad -> conflict-free 16B frag loads)
#define FS   68        // f32 stage row stride

typedef _Float16 half8  __attribute__((ext_vector_type(8)));
typedef _Float16 half16 __attribute__((ext_vector_type(16)));
typedef float    float8 __attribute__((ext_vector_type(8)));

// Load one wave-fragment of a 16x32 f16 operand from LDS.
// Lane layout (A: m = lane%16, B: n = lane%16; hi = lane/16):
//   halves [0..7]  = row[kk + hi*8 + 0..7]
//   halves [8..15] = row[kk + hi*8 + 16 + 0..7]
__device__ __forceinline__ half16 ldsfrag(const _Float16* rowp, int kk, int hi) {
  const half8 a = *(const half8*)(rowp + kk + hi * 8);
  const half8 b = *(const half8*)(rowp + kk + hi * 8 + 16);
  half16 r;
#pragma unroll
  for (int i = 0; i < 8; ++i) { r[i] = a[i]; r[i + 8] = b[i]; }
  return r;
}

// CDNA5 async global->LDS copy of 16 bytes per lane (ASYNCcnt-tracked, no VGPR data).
// lds_addr: per-lane LDS byte address (low 32 bits of flat shared address).
// gbase:    uniform 64-bit base (SGPR pair), goff: per-lane signed byte offset.
__device__ __forceinline__ void async_cp16(unsigned lds_addr, const void* gbase,
                                           unsigned goff) {
  asm volatile("global_load_async_to_lds_b128 %0, %1, %2 offset:0"
               :: "v"(lds_addr), "v"(goff), "s"(gbase)
               : "memory");
}
__device__ __forceinline__ void wait_async_le8() {
  asm volatile("s_wait_asynccnt 0x8" ::: "memory");
}
__device__ __forceinline__ void wait_async_le4() {
  asm volatile("s_wait_asynccnt 0x4" ::: "memory");
}
__device__ __forceinline__ void wait_async_0() {
  asm volatile("s_wait_asynccnt 0x0" ::: "memory");
}

// ---------------------------------------------------------------------------
// K0: fold Ws into Wx:  A[h*64+g, k] = sum_d Ws[g,d] * Wx[h*64+d, k]
//     c[h*64+g] = bs[g] + sum_d Ws[g,d] * bx[h*64+d]
// ---------------------------------------------------------------------------
__global__ __launch_bounds__(64) void k0_prep(const float* __restrict__ Ws,
                                              const float* __restrict__ bs,
                                              const float* __restrict__ Wx,
                                              const float* __restrict__ bx,
                                              float* __restrict__ Aw,
                                              float* __restrict__ cw) {
  __shared__ float wsr[64];
  const int hg = blockIdx.x;
  const int h = hg >> 6, g = hg & 63;
  const int t = threadIdx.x;
  wsr[t] = Ws[g * 64 + t];
  __syncthreads();
  for (int k = t; k < 512; k += 64) {
    float a = 0.f;
#pragma unroll 4
    for (int d = 0; d < 64; ++d) a += wsr[d] * Wx[(h * 64 + d) * 512 + k];
    Aw[hg * 512 + k] = a;
  }
  if (t == 0) {
    float c = bs[g];
    for (int d = 0; d < 64; ++d) c += wsr[d] * bx[h * 64 + d];
    cw[hg] = c;
  }
}

// ---------------------------------------------------------------------------
// K1: per (split s, head h). LDS-resident 128x512 f16 weight slab
//     (rows 0..63 = Wfx head slice, rows 64..127 = A head slice).
//     x chunks stream via double-buffered async global->LDS copies, are
//     converted f32->f16 into the padded operand slab, then WMMA -> fx &
//     logits, softmax(T*logits) -> w (f16 to global), pooling tok/norm in LDS,
//     per-block partials flushed at the end (no atomics).
// ---------------------------------------------------------------------------
__global__ __launch_bounds__(256) void k1_main(const float* __restrict__ x,
    const float* __restrict__ Wfx, const float* __restrict__ bfx,
    const float* __restrict__ Aw, const float* __restrict__ cw,
    const float* __restrict__ temp,
    _Float16* __restrict__ wflat, float* __restrict__ ptok,
    float* __restrict__ pnorm, int N) {
  extern __shared__ __align__(16) char smem[];
  _Float16* wslab = (_Float16*)smem;          // 128 * LSTR f16
  _Float16* xs    = wslab + 128 * LSTR;       // 16 * LSTR f16
  float* tok = (float*)(xs + 16 * LSTR);      // 64*64 f32
  float* nrm = tok + 64 * 64;                 // 64 f32
  float* fxs = nrm + 64;                      // 16*FS f32
  float* ls  = fxs + 16 * FS;                 // 16*FS f32
  float* raw0 = ls + 16 * FS;                 // 16*512 f32 (async landing, buf 0)
  float* raw1 = raw0 + 16 * 512;              // 16*512 f32 (buf 1)

  const int s = blockIdx.x, h = blockIdx.y;
  const int t = threadIdx.x;
  const int wv = t >> 5, ln = t & 31, l15 = ln & 15, hi = ln >> 4;
  const unsigned raw_lds[2] = {(unsigned)(uintptr_t)raw0,
                               (unsigned)(uintptr_t)raw1};

  // weight slab (f32 -> f16)
  for (int e = t; e < 128 * 512; e += 256) {
    int r = e >> 9, k = e & 511;
    float v = (r < 64) ? Wfx[(h * 64 + r) * 512 + k]
                       : Aw[(h * 64 + (r - 64)) * 512 + k];
    wslab[r * LSTR + k] = (_Float16)v;
  }
  for (int e = t; e < 64 * 64; e += 256) tok[e] = 0.f;
  if (t < 64) nrm[t] = 0.f;

  const int rows_per = (N + S1 - 1) / S1;
  const int base = s * rows_per;
  const int count = min(rows_per, N - base);
  const int nch = (count + 15) >> 4;
  const float T = temp[h];

  const _Float16* arow = xs + l15 * LSTR;
  const _Float16* brow =
      wslab + (((wv < 4) ? (wv * 16) : (64 + (wv - 4) * 16)) + l15) * LSTR;

  // issue async copy of one 16x512 f32 x-chunk (8 x 16B per thread)
  auto issue_x = [&](int rbase, unsigned lds_base) {
#pragma unroll
    for (int j = 0; j < 8; ++j) {
      int q = t + 256 * j;          // 16B chunk id, 0..2047
      int r = q >> 7;               // row 0..15
      int k4 = (q & 127) << 2;      // f32 col 0..508
      int row = rbase + r;
      row = (row < N) ? row : (N - 1);  // clamp: never OOB (tail zeroed later)
      unsigned goff = ((unsigned)row * 512u + (unsigned)k4) * 4u;
      async_cp16(lds_base + (unsigned)q * 16u, (const void*)x, goff);
    }
  };

  issue_x(base, raw_lds[0]);        // prologue prefetch of chunk 0
  __syncthreads();                  // wslab/tok/nrm ready

  for (int ch = 0; ch < nch; ++ch) {
    const int rbase = base + (ch << 4);
    const int rem = count - (ch << 4);  // valid rows this chunk (may exceed 16)

    if (ch + 1 < nch) {
      issue_x(base + ((ch + 1) << 4), raw_lds[(ch + 1) & 1]);
      wait_async_le8();             // in-order: chunk ch landed
    } else {
      wait_async_0();
    }
    __syncthreads();

    // convert landed f32 chunk -> padded f16 operand slab (zero tail rows)
    {
      const float* rawb = (ch & 1) ? raw1 : raw0;
      for (int e = t; e < 16 * 512; e += 256) {
        int r = e >> 9, k = e & 511;
        xs[r * LSTR + k] = (r < rem) ? (_Float16)rawb[e] : (_Float16)0.f;
      }
    }
    __syncthreads();

    float8 acc = {0.f, 0.f, 0.f, 0.f, 0.f, 0.f, 0.f, 0.f};
#pragma unroll
    for (int kk = 0; kk < 512; kk += 32) {
      half16 af = ldsfrag(arow, kk, hi);
      half16 bf = ldsfrag(brow, kk, hi);
      acc = __builtin_amdgcn_wmma_f32_16x16x32_f16(false, af, false, bf,
                                                   (short)0, acc, false, false);
    }
    if (wv < 4) {  // fx tile, cols d = wv*16 + l15
      const int dcol = wv * 16 + l15;
      const float bias = bfx[h * 64 + dcol];
#pragma unroll
      for (int v = 0; v < 8; ++v) fxs[(v + 8 * hi) * FS + dcol] = acc[v] + bias;
    } else {       // logits tile, cols g = (wv-4)*16 + l15
      const int gcol = (wv - 4) * 16 + l15;
      const float bias = cw[h * 64 + gcol];
#pragma unroll
      for (int v = 0; v < 8; ++v) ls[(v + 8 * hi) * FS + gcol] = acc[v] + bias;
    }
    __syncthreads();

    // softmax over G=64, 16 threads per row
    {
      const int r = t >> 4, cj = t & 15;
      float lv[4];
#pragma unroll
      for (int j = 0; j < 4; ++j) lv[j] = ls[r * FS + cj + 16 * j] * T;
      float mx = fmaxf(fmaxf(lv[0], lv[1]), fmaxf(lv[2], lv[3]));
#pragma unroll
      for (int m = 1; m < 16; m <<= 1) mx = fmaxf(mx, __shfl_xor(mx, m, 16));
      float ev[4]; float sm = 0.f;
#pragma unroll
      for (int j = 0; j < 4; ++j) { ev[j] = __expf(lv[j] - mx); sm += ev[j]; }
#pragma unroll
      for (int m = 1; m < 16; m <<= 1) sm += __shfl_xor(sm, m, 16);
      const float inv = 1.f / sm;
      const bool valid = (r < rem);
#pragma unroll
      for (int j = 0; j < 4; ++j) {
        const float w = valid ? ev[j] * inv : 0.f;
        ls[r * FS + cj + 16 * j] = w;
        if (valid)
          wflat[(size_t)(rbase + r) * 512 + h * 64 + cj + 16 * j] = (_Float16)w;
      }
    }
    __syncthreads();

    // pooling: tok[g][d] += sum_r w[r][g]*fx[r][d]; norm[g] += sum_r w[r][g]
    {
      const int d = t & 63, gq = t >> 6;  // each thread owns g in {gq+4j}, one d
      float pa[16];
#pragma unroll
      for (int j = 0; j < 16; ++j) pa[j] = 0.f;
      for (int r = 0; r < 16; ++r) {
        const float fv = fxs[r * FS + d];
#pragma unroll
        for (int j = 0; j < 16; ++j) pa[j] += fv * ls[r * FS + gq + 4 * j];
      }
#pragma unroll
      for (int j = 0; j < 16; ++j) tok[(gq + 4 * j) * 64 + d] += pa[j];
      if (t < 64) {
        float sn = 0.f;
        for (int r = 0; r < 16; ++r) sn += ls[r * FS + t];
        nrm[t] += sn;
      }
    }
    __syncthreads();
  }

  float* pt = ptok + (size_t)(s * HH + h) * 4096;
  for (int e = t; e < 4096; e += 256) pt[e] = tok[e];
  if (t < 64) pnorm[(s * HH + h) * 64 + t] = nrm[t];
}

// ---------------------------------------------------------------------------
// K2: one block per head. Reduce pooling partials, normalize, tiny G x G
//     attention, then fold Wo and Wout into Mt[o2][h*64+g] (f16, transposed).
// ---------------------------------------------------------------------------
__global__ __launch_bounds__(256) void k2_head(const float* __restrict__ ptok,
    const float* __restrict__ pnorm,
    const float* __restrict__ Wq, const float* __restrict__ Wk,
    const float* __restrict__ Wv, const float* __restrict__ Wo,
    const float* __restrict__ Wout, _Float16* __restrict__ Mt) {
  extern __shared__ __align__(16) char smem[];
  float* tokb = (float*)smem;       // 64*65
  float* qb = tokb + 64 * 65;
  float* kb = qb + 64 * 65;
  float* vb = kb + 64 * 65;
  float* nr = vb + 64 * 65;         // 64
  const int h = blockIdx.x, t = threadIdx.x;

  if (t < 64) {
    float a = 0.f;
    for (int s = 0; s < S1; ++s) a += pnorm[(s * HH + h) * 64 + t];
    nr[t] = a + EPSF;
  }
  __syncthreads();
  for (int e = t; e < 4096; e += 256) {
    int g = e >> 6, d = e & 63;
    float a = 0.f;
    for (int s = 0; s < S1; ++s) a += ptok[(size_t)(s * HH + h) * 4096 + e];
    tokb[g * 65 + d] = a / nr[g];
  }
  __syncthreads();
  for (int e = t; e < 4096; e += 256) {
    int g = e >> 6, o = e & 63;
    float aq = 0.f, ak = 0.f, av = 0.f;
    for (int d = 0; d < 64; ++d) {
      float td = tokb[g * 65 + d];
      aq += td * Wq[o * 64 + d];
      ak += td * Wk[o * 64 + d];
      av += td * Wv[o * 64 + d];
    }
    qb[g * 65 + o] = aq; kb[g * 65 + o] = ak; vb[g * 65 + o] = av;
  }
  __syncthreads();
  if (t < 64) {  // attention row t
    float sc[64];
    float mx = -1e30f;
    for (int j = 0; j < 64; ++j) {
      float a = 0.f;
      for (int d = 0; d < 64; ++d) a += qb[t * 65 + d] * kb[j * 65 + d];
      a *= 0.125f;  // 1/sqrt(64)
      sc[j] = a; mx = fmaxf(mx, a);
    }
    float sm = 0.f;
    for (int j = 0; j < 64; ++j) { sc[j] = __expf(sc[j] - mx); sm += sc[j]; }
    const float inv = 1.f / sm;
    for (int d = 0; d < 64; ++d) {
      float a = 0.f;
      for (int j = 0; j < 64; ++j) a += sc[j] * vb[j * 65 + d];
      tokb[t * 65 + d] = a * inv;  // out_tok (reuse tokb)
    }
  }
  __syncthreads();
  for (int e = t; e < 4096; e += 256) {  // tmp = out_tok @ Wo^T -> qb
    int g = e >> 6, o = e & 63;
    float a = 0.f;
    for (int d = 0; d < 64; ++d) a += tokb[g * 65 + d] * Wo[o * 64 + d];
    qb[g * 65 + o] = a;
  }
  __syncthreads();
  for (int e = t; e < 512 * 64; e += 256) {  // Mt[o2][hg] = tmp @ WoutSlice^T
    int g = e & 63, o2 = e >> 6;
    float a = 0.f;
    for (int o = 0; o < 64; ++o)
      a += qb[g * 65 + o] * Wout[o2 * 512 + (h << 6) + o];
    Mt[(size_t)o2 * 512 + (h << 6) + g] = (_Float16)a;
  }
}

// ---------------------------------------------------------------------------
// K3: out = w_flat(N x 512, f16) @ M(512 x 512) + bout.
//     Mt strip and w chunks staged entirely with async global->LDS copies
//     (f16 -> f16, landing directly in the padded operand layout).
// ---------------------------------------------------------------------------
__global__ __launch_bounds__(256) void k3_out(const _Float16* __restrict__ wflat,
    const _Float16* __restrict__ Mt, const float* __restrict__ bout,
    float* __restrict__ out, int N) {
  extern __shared__ __align__(16) char smem[];
  _Float16* mts  = (_Float16*)smem;       // 128 * LSTR
  _Float16* wst0 = mts + 128 * LSTR;      // 16 * LSTR (buf 0)
  _Float16* wst1 = wst0 + 16 * LSTR;      // 16 * LSTR (buf 1)
  const int strip = blockIdx.x, s = blockIdx.y;
  const int t = threadIdx.x;
  const int wv = t >> 5, ln = t & 31, l15 = ln & 15, hi = ln >> 4;
  const int CB = strip * 128;
  const unsigned mts_lds = (unsigned)(uintptr_t)mts;
  const unsigned wst_lds[2] = {(unsigned)(uintptr_t)wst0,
                               (unsigned)(uintptr_t)wst1};

  // async slab load: Mt rows CB..CB+127 into padded LDS (32 x 16B per thread)
#pragma unroll
  for (int j = 0; j < 32; ++j) {
    int q = t + 256 * j;            // 8-half chunk id, 0..8191
    int r = q >> 6;                 // slab row 0..127
    int kh = (q & 63) << 3;         // half col 0..504
    unsigned goff = (((unsigned)(CB + r) * 512u) + (unsigned)kh) * 2u;
    async_cp16(mts_lds + ((unsigned)r * LSTR + (unsigned)kh) * 2u,
               (const void*)Mt, goff);
  }

  const int rows_per = (N + R3 - 1) / R3;
  const int base = s * rows_per;
  const int count = min(rows_per, N - base);
  const int nch = (count + 15) >> 4;
  const int col = CB + wv * 16 + l15;
  const float bb = bout[col];
  const _Float16* arow0 = wst0 + l15 * LSTR;
  const _Float16* arow1 = wst1 + l15 * LSTR;
  const _Float16* brow = mts + (wv * 16 + l15) * LSTR;

  // issue async copy of one 16x512 f16 w-chunk (4 x 16B per thread)
  auto issue_w = [&](int rbase, unsigned lds_base) {
#pragma unroll
    for (int j = 0; j < 4; ++j) {
      int q = t + 256 * j;          // 0..1023
      int r = q >> 6;               // row 0..15
      int kh = (q & 63) << 3;       // half col
      int row = rbase + r;
      row = (row < N) ? row : (N - 1);  // clamp (tail rows masked at store)
      unsigned goff = ((unsigned)row * 512u + (unsigned)kh) * 2u;
      async_cp16(lds_base + ((unsigned)r * LSTR + (unsigned)kh) * 2u,
                 (const void*)wflat, goff);
    }
  };

  issue_w(base, wst_lds[0]);
  __syncthreads();

  for (int ch = 0; ch < nch; ++ch) {
    const int rbase = base + (ch << 4);
    const int rem = count - (ch << 4);

    if (ch + 1 < nch) {
      issue_w(base + ((ch + 1) << 4), wst_lds[(ch + 1) & 1]);
      wait_async_le4();             // in-order: slab + chunk ch landed
    } else {
      wait_async_0();
    }
    __syncthreads();

    float8 acc = {0.f, 0.f, 0.f, 0.f, 0.f, 0.f, 0.f, 0.f};
    const _Float16* arow = (ch & 1) ? arow1 : arow0;
#pragma unroll
    for (int kk = 0; kk < 512; kk += 32) {
      half16 af = ldsfrag(arow, kk, hi);
      half16 bf = ldsfrag(brow, kk, hi);
      acc = __builtin_amdgcn_wmma_f32_16x16x32_f16(false, af, false, bf,
                                                   (short)0, acc, false, false);
    }
#pragma unroll
    for (int v = 0; v < 8; ++v) {
      int r = v + 8 * hi;
      if (r < rem) out[(size_t)(rbase + r) * 512 + col] = acc[v] + bb;
    }
    __syncthreads();
  }
}

// ---------------------------------------------------------------------------
extern "C" void kernel_launch(void* const* d_in, const int* in_sizes, int n_in,
                              void* d_out, int out_size, void* d_ws, size_t ws_size,
                              hipStream_t stream) {
  const float* x    = (const float*)d_in[0];
  const float* Wx   = (const float*)d_in[1];
  const float* bx   = (const float*)d_in[2];
  const float* Wfx  = (const float*)d_in[3];
  const float* bfx  = (const float*)d_in[4];
  const float* Ws   = (const float*)d_in[5];
  const float* bs   = (const float*)d_in[6];
  const float* temp = (const float*)d_in[7];
  const float* Wq   = (const float*)d_in[8];
  const float* Wk   = (const float*)d_in[9];
  const float* Wv   = (const float*)d_in[10];
  const float* Wo   = (const float*)d_in[11];
  const float* Wout = (const float*)d_in[12];
  const float* bout = (const float*)d_in[13];
  float* out = (float*)d_out;
  const int N = in_sizes[0] / HID2;
  (void)n_in; (void)out_size; (void)ws_size;

  char* p = (char*)d_ws;
  auto wsalloc = [&](size_t bytes) -> char* {
    char* q = p;
    p += (bytes + 255) & ~(size_t)255;
    return q;
  };
  float*    Aw    = (float*)wsalloc((size_t)512 * 512 * 4);
  float*    cw    = (float*)wsalloc(512 * 4);
  float*    ptok  = (float*)wsalloc((size_t)S1 * HH * 4096 * 4);
  float*    pnorm = (float*)wsalloc((size_t)S1 * HH * 64 * 4);
  _Float16* Mt    = (_Float16*)wsalloc((size_t)512 * 512 * 2);
  _Float16* wflat = (_Float16*)wsalloc((size_t)N * 512 * 2);

  const size_t sm1 = (size_t)(128 + 16) * LSTR * 2 +
                     (size_t)(64 * 64 + 64 + 2 * 16 * FS) * 4 +
                     (size_t)2 * 16 * 512 * 4;                  // ~235 KB
  const size_t sm2 = (size_t)(4 * 64 * 65 + 64) * 4;            // ~65 KB
  const size_t sm3 = (size_t)(128 + 2 * 16) * LSTR * 2;         // ~163 KB
  hipFuncSetAttribute(reinterpret_cast<const void*>(k1_main),
                      hipFuncAttributeMaxDynamicSharedMemorySize, (int)sm1);
  hipFuncSetAttribute(reinterpret_cast<const void*>(k2_head),
                      hipFuncAttributeMaxDynamicSharedMemorySize, (int)sm2);
  hipFuncSetAttribute(reinterpret_cast<const void*>(k3_out),
                      hipFuncAttributeMaxDynamicSharedMemorySize, (int)sm3);

  k0_prep<<<512, 64, 0, stream>>>(Ws, bs, Wx, bx, Aw, cw);
  k1_main<<<dim3(S1, HH), 256, sm1, stream>>>(x, Wfx, bfx, Aw, cw, temp,
                                              wflat, ptok, pnorm, N);
  k2_head<<<HH, 256, sm2, stream>>>(ptok, pnorm, Wq, Wk, Wv, Wo, Wout, Mt);
  k3_out<<<dim3(4, R3), 256, sm3, stream>>>(wflat, Mt, bout, out, N);
}
